// Attention_652835029261
// MI455X (gfx1250) — matrix operand order repaired and checked by reading, not compiled, not run on hardware
//
#include <hip/hip_runtime.h>

// ---------------------------------------------------------------------------
// Sigmoid "fast attention" for MI455X (gfx1250, wave32, WMMA + async-to-LDS).
//   q = (x@Wq)*scale + bias ; k = x@Wk ; v = reshape(x)
//   out = sigmoid(q k^T) v          (no softmax -> pure streaming accumulate)
// f16 operands, f32 WMMA accumulation via v_wmma_f32_16x16x32_f16.
// K / V^T tiles double-buffered in LDS via GLOBAL_LOAD_ASYNC_TO_LDS (ASYNCcnt).
// ---------------------------------------------------------------------------

typedef __attribute__((ext_vector_type(16))) _Float16 v16h;
typedef __attribute__((ext_vector_type(8)))  float    v8f;

struct alignas(16) h8 { _Float16 h[8]; };
union AFrag { v16h v; h8 p[2]; };

#define B_     4
#define N_     2048
#define DIM_   512
#define HEADS_ 8
#define DK_    64
#define SCALE_ 0.125f   // 64^-0.5

// ---- gfx1250 async global->LDS copy (ASYNCcnt), with safe fallbacks --------
#if defined(__has_builtin)
#if __has_builtin(__builtin_amdgcn_global_load_async_to_lds_b128)
#define HAVE_ASYNC_LDS 1
#endif
#endif

// builtin signature (from hipcc diagnostic): param0 = vector_size(16) int,
// device (global) address space, non-const; param1 = LDS address space.
typedef int v4i_vs __attribute__((vector_size(4 * sizeof(int))));
typedef __attribute__((address_space(1))) v4i_vs* g_v4i_ptr;
typedef __attribute__((address_space(3))) v4i_vs* l_v4i_ptr;

__device__ __forceinline__ void async_copy_b128(const _Float16* g, _Float16* l) {
#if defined(HAVE_ASYNC_LDS)
    __builtin_amdgcn_global_load_async_to_lds_b128(
        (g_v4i_ptr)(_Float16*)g, (l_v4i_ptr)l, 0, 0);
#else
    *(h8*)l = *(const h8*)g;   // synchronous fallback
#endif
}

__device__ __forceinline__ void wait_async0() {
#if defined(HAVE_ASYNC_LDS)
#if __has_builtin(__builtin_amdgcn_s_wait_asynccnt)
    __builtin_amdgcn_s_wait_asynccnt(0);
#else
    asm volatile("s_wait_asynccnt 0" ::: "memory");
#endif
#endif
}

// ===========================================================================
// Pass 1: C[8192x512] = X @ W, fused scale+bias for Q, output f16 [b,h,n,64]
// 256 threads (8 waves); block computes 128 rows x 64 cols; K-chunks of 32.
// (staging stays synchronous: it fuses the f32->f16 convert + W transpose)
// ===========================================================================
__global__ __launch_bounds__(256) void proj_kernel(
    const float* __restrict__ X, const float* __restrict__ W,
    const float* __restrict__ bias, _Float16* __restrict__ Out, int is_q)
{
    __shared__ alignas(16) _Float16 sX [128 * 32]; // A tiles, row-major
    __shared__ alignas(16) _Float16 sWT[ 64 * 32]; // B tiles, transposed [col][k]

    const int tid  = threadIdx.x;
    const int lane = tid & 31;
    const int wave = tid >> 5;
    const int bm0  = blockIdx.x * 128;   // row block of 8192 (= b*2048+n)
    const int bn0  = blockIdx.y * 64;    // col block of 512

    v8f acc[4] = {};

    for (int kc = 0; kc < DIM_; kc += 32) {
        __syncthreads();
        // --- stage X chunk (128x32 f32 -> f16, row-major) ---
        {
            int row = tid >> 1;
            int c0  = (tid & 1) * 16;
            const float* src = X + (size_t)(bm0 + row) * DIM_ + kc + c0;
            if (kc + 32 < DIM_) __builtin_prefetch(src + 32, 0, 0);
            _Float16* dst = &sX[row * 32 + c0];
            #pragma unroll
            for (int i = 0; i < 16; ++i) dst[i] = (_Float16)src[i];
        }
        // --- stage W chunk transposed (32x64 -> sWT[col][k]) ---
        {
            int idx = tid * 8;            // 0..2047
            int kr  = idx >> 6;           // k-row within chunk 0..31
            int c0  = idx & 63;
            const float* src = W + (size_t)(kc + kr) * DIM_ + bn0 + c0;
            if (kc + 32 < DIM_) __builtin_prefetch(src + 32 * DIM_, 0, 0);
            #pragma unroll
            for (int i = 0; i < 8; ++i) sWT[(c0 + i) * 32 + kr] = (_Float16)src[i];
        }
        __syncthreads();

        // --- A fragment: 16 rows of this wave, K = kc..kc+31 ---
        AFrag a;
        {
            int m  = lane & 15;
            int kb = (lane < 16) ? 0 : 8;
            const _Float16* r = &sX[(wave * 16 + m) * 32];
            a.p[0] = *(const h8*)(r + kb);
            a.p[1] = *(const h8*)(r + kb + 16);
        }
        // --- 4 N-tiles of 16 cols each ---
        #pragma unroll
        for (int nt = 0; nt < 4; ++nt) {
            AFrag bf;
            int c  = nt * 16 + (lane & 15);
            int rb = (lane < 16) ? 0 : 8;
            bf.p[0] = *(const h8*)&sWT[c * 32 + rb];
            bf.p[1] = *(const h8*)&sWT[c * 32 + rb + 16];
            acc[nt] = __builtin_amdgcn_wmma_f32_16x16x32_f16(
                false, a.v, false, bf.v, (short)0, acc[nt], false, false);
        }
    }

    // --- epilogue: scale+bias (Q), f16 store in [b,h,n,64] layout ---
    const int nlo = lane & 15;
    const int mb  = (lane < 16) ? 0 : 8;
    #pragma unroll
    for (int nt = 0; nt < 4; ++nt) {
        int cg = bn0 + nt * 16 + nlo;          // 0..511
        int h  = cg >> 6;
        int d  = cg & 63;
        float bv = is_q ? bias[h * DK_ + d] : 0.0f;
        #pragma unroll
        for (int r = 0; r < 8; ++r) {
            int rg = bm0 + wave * 16 + mb + r; // global row = b*2048+n
            int b  = rg >> 11;
            int n  = rg & (N_ - 1);
            float val = acc[nt][r];
            if (is_q) val = val * SCALE_ + bv;
            Out[(((size_t)(b * HEADS_ + h)) * N_ + n) * DK_ + d] = (_Float16)val;
        }
    }
}

// ===========================================================================
// Pass 2: vt[b,h,d,n] = (f16) x[b,n,h*64+d]   (V transposed for async staging)
// ===========================================================================
__global__ __launch_bounds__(256) void vconv(
    const float* __restrict__ X, _Float16* __restrict__ VT)
{
    size_t t  = (size_t)blockIdx.x * blockDim.x + threadIdx.x;
    size_t i  = t * 8;                 // flat index into vt[bh][d][n]
    int n0    = (int)(i & (N_ - 1));
    size_t r  = i >> 11;
    int d     = (int)(r & (DK_ - 1));
    int bh    = (int)(r >> 6);
    int h     = bh & (HEADS_ - 1);
    int b     = bh >> 3;
    const float* src = X + ((size_t)(b * N_ + n0)) * DIM_ + h * DK_ + d;
    h8 v;
    #pragma unroll
    for (int j = 0; j < 8; ++j) v.h[j] = (_Float16)src[(size_t)j * DIM_];
    *(h8*)(VT + i) = v;
}

// ===========================================================================
// Pass 3: streaming sigmoid attention. Block = (b,h) x 128 query rows.
// 8 waves x 16 rows; key blocks of 32; double-buffered async K/V^T staging.
// ===========================================================================
__global__ __launch_bounds__(256) void attn_kernel(
    const _Float16* __restrict__ Q, const _Float16* __restrict__ K,
    const _Float16* __restrict__ VT, float* __restrict__ Out)
{
    __shared__ alignas(16) _Float16 sK [2][32 * 64]; // K tile [key][dim]
    __shared__ alignas(16) _Float16 sVT[2][64 * 32]; // V^T tile [dim][key]
    __shared__ alignas(16) _Float16 sP [8][16 * 32]; // per-wave P staging

    const int tid  = threadIdx.x;
    const int lane = tid & 31;
    const int wave = tid >> 5;
    const int bh   = blockIdx.y;                    // 0..31
    const int m0   = blockIdx.x * 128;              // query row block

    const size_t base = (size_t)bh * N_ * DK_;      // also VT base (64*2048)
    const int nlo = lane & 15;
    const int rb  = (lane < 16) ? 0 : 16;
    const int kb  = (lane < 16) ? 0 : 8;
    const int mb  = (lane < 16) ? 0 : 8;

    // per-thread staging coordinates (one b128 each for K and VT per tile)
    const int cidx = tid * 8;          // 0..2047 halves
    const int vd   = cidx >> 5;        // VT row (dim)  0..63
    const int vn   = cidx & 31;        // VT col (key)  0,8,16,24

    // --- Q fragments (16 rows x 64 dims) held in registers for all key tiles ---
    AFrag qf[2];
    {
        const _Float16* qrow = Q + base + (size_t)(m0 + wave * 16 + nlo) * DK_;
        qf[0].p[0] = *(const h8*)(qrow + kb);
        qf[0].p[1] = *(const h8*)(qrow + kb + 16);
        qf[1].p[0] = *(const h8*)(qrow + 32 + kb);
        qf[1].p[1] = *(const h8*)(qrow + 32 + kb + 16);
    }

    v8f oacc[4] = {};

    // prologue: issue async copies for key tile 0 into buffer 0
    async_copy_b128(K  + base + cidx,                         &sK [0][cidx]);
    async_copy_b128(VT + base + (size_t)vd * N_ + 0 + vn,     &sVT[0][cidx]);

    for (int jt = 0; jt < N_ / 32; ++jt) {
        const int buf = jt & 1;
        wait_async0();        // this wave's copies for tile jt complete
        __syncthreads();      // => all waves' copies complete; prev reads done

        // issue async copies for the next tile into the other buffer
        if (jt + 1 < N_ / 32) {
            const int j1 = (jt + 1) * 32;
            async_copy_b128(K  + base + (size_t)j1 * DK_ + cidx,   &sK [buf ^ 1][cidx]);
            async_copy_b128(VT + base + (size_t)vd * N_ + j1 + vn, &sVT[buf ^ 1][cidx]);
        }

        // --- S = Q K^T : two 16x16 tiles, dk=64 as two K=32 WMMAs each ---
        v8f s[2];
        #pragma unroll
        for (int jn = 0; jn < 2; ++jn) {
            v8f sa = {};
            #pragma unroll
            for (int kc = 0; kc < 2; ++kc) {
                AFrag bf;
                const _Float16* kr = &sK[buf][(jn * 16 + nlo) * 64 + kc * 32 + kb];
                bf.p[0] = *(const h8*)(kr);
                bf.p[1] = *(const h8*)(kr + 16);
                sa = __builtin_amdgcn_wmma_f32_16x16x32_f16(
                    false, qf[kc].v, false, bf.v, (short)0, sa, false, false);
            }
            s[jn] = sa;
        }

        // --- sigmoid (f32) and stage P (f16) for A-layout reload ---
        #pragma unroll
        for (int jn = 0; jn < 2; ++jn) {
            #pragma unroll
            for (int r = 0; r < 8; ++r) {
                float xv = s[jn][r];
                float p  = 1.0f / (1.0f + __expf(-xv));
                sP[wave][(mb + r) * 32 + jn * 16 + nlo] = (_Float16)p;
            }
        }
        __syncthreads();

        // --- O += P @ V : 4 output tiles of 16 dims ---
        AFrag pf;
        pf.p[0] = *(const h8*)&sP[wave][nlo * 32 + kb];
        pf.p[1] = *(const h8*)&sP[wave][nlo * 32 + kb + 16];
        #pragma unroll
        for (int nt = 0; nt < 4; ++nt) {
            AFrag vf;
            const _Float16* vr = &sVT[buf][(nt * 16 + nlo) * 32 + kb];
            vf.p[0] = *(const h8*)(vr);
            vf.p[1] = *(const h8*)(vr + 16);
            oacc[nt] = __builtin_amdgcn_wmma_f32_16x16x32_f16(
                false, pf.v, false, vf.v, (short)0, oacc[nt], false, false);
        }
    }

    // --- write out[b][n][h*64+d] (f32) ---
    {
        int h = bh & (HEADS_ - 1);
        int b = bh >> 3;
        #pragma unroll
        for (int nt = 0; nt < 4; ++nt) {
            int d = nt * 16 + nlo;
            #pragma unroll
            for (int r = 0; r < 8; ++r) {
                int n = m0 + wave * 16 + mb + r;
                Out[((size_t)(b * N_ + n)) * DIM_ + h * DK_ + d] = oacc[nt][r];
            }
        }
    }
}

// ===========================================================================
extern "C" void kernel_launch(void* const* d_in, const int* in_sizes, int n_in,
                              void* d_out, int out_size, void* d_ws, size_t ws_size,
                              hipStream_t stream)
{
    const float* x    = (const float*)d_in[0];
    const float* Wq   = (const float*)d_in[1];
    const float* Wk   = (const float*)d_in[2];
    const float* bias = (const float*)d_in[3];
    float* out = (float*)d_out;

    const size_t per = (size_t)B_ * HEADS_ * N_ * DK_;   // 4,194,304 halves
    _Float16* qf  = (_Float16*)d_ws;
    _Float16* kf  = qf + per;
    _Float16* vtf = kf + per;

    dim3 pg((B_ * N_) / 128, DIM_ / 64);                 // 64 x 8
    proj_kernel<<<pg, 256, 0, stream>>>(x, Wq, bias, qf, 1);
    proj_kernel<<<pg, 256, 0, stream>>>(x, Wk, bias, kf, 0);

    vconv<<<(unsigned)((per / 8 + 255) / 256), 256, 0, stream>>>(x, vtf);

    dim3 ag(N_ / 128, B_ * HEADS_);                      // 16 x 32
    attn_kernel<<<ag, 256, 0, stream>>>(qf, kf, vtf, out);
}
